// LSTM_62251255989030
// MI455X (gfx1250) — matrix operand order repaired and checked by reading, not compiled, not run on hardware
//
#include <hip/hip_runtime.h>
#include <stdint.h>

// Problem dims (fixed by reference)
#define N_   128
#define L_   2048
#define H_   512
#define G4_  (4 * H_)          // 2048 gate columns
#define NWG  32                // persistent workgroups
#define TPB  256               // 8 wave32 per WG
#define WCOLS (H_ / NWG)       // 16 H-columns owned per WG -> 4 gate tiles/wave

typedef __attribute__((ext_vector_type(16))) __bf16 v16bf;
typedef __attribute__((ext_vector_type(8)))  float  v8f;

union FragBF { struct { uint4 lo, hi; } q; v16bf v; };

__device__ __forceinline__ unsigned short f2bf(float f) {
  union { float f; unsigned u; } v; v.f = f;
  unsigned r = v.u + 0x7FFFu + ((v.u >> 16) & 1u);   // round-to-nearest-even
  return (unsigned short)(r >> 16);
}
__device__ __forceinline__ float bf2f(unsigned short b) {
  union { unsigned u; float f; } v; v.u = (unsigned)b << 16; return v.f;
}
__device__ __forceinline__ float sigmoidf_(float x) { return 1.0f / (1.0f + __expf(-x)); }

// ---------------- init kernels ----------------
__global__ void cvt_bf16_kernel(const float* __restrict__ src,
                                unsigned short* __restrict__ dst, int n) {
  int i = blockIdx.x * blockDim.x + threadIdx.x;
  if (i < n) dst[i] = f2bf(src[i]);
}

__global__ void init_state_kernel(unsigned short* __restrict__ h1,
                                  unsigned short* __restrict__ h2,
                                  unsigned* __restrict__ cnt) {
  int i = blockIdx.x * blockDim.x + threadIdx.x;
  const int total = 2 * N_ * H_;   // double-buffered
  if (i < total) { h1[i] = 0; h2[i] = 0; }
  if (i == 0) *cnt = 0;
}

// ---------------- grid-wide barrier (monotonic counter) ----------------
__device__ __forceinline__ void grid_barrier(unsigned* cnt, unsigned idx) {
  __syncthreads();
  if (threadIdx.x == 0) {
    __threadfence();  // release h writes to device scope
    __hip_atomic_fetch_add(cnt, 1u, __ATOMIC_ACQ_REL, __HIP_MEMORY_SCOPE_AGENT);
    const unsigned target = idx * NWG;
    while (__hip_atomic_load(cnt, __ATOMIC_ACQUIRE, __HIP_MEMORY_SCOPE_AGENT) < target) {
      __builtin_amdgcn_s_sleep(2);
    }
  }
  __syncthreads();
}

// ---------------- WMMA GEMM: acc[g] += A(h)[16 rows x K] * W^T tile(g) ----------------
// A: bf16 [N_,H_] row-major (rows = batch). W: bf16 [G4_,H_] row-major (B = W^T).
// Wave computes rows [rowBase, rowBase+16), gate columns g*H_ + colBase + 0..15.
__device__ __forceinline__ void gemm_acc(v8f* acc,
                                         const unsigned short* __restrict__ A,
                                         const unsigned short* __restrict__ W,
                                         int rowBase, int colBase,
                                         int l15, int hiHalf) {
  const int aKoff = hiHalf * 8;    // elements
  const int bKoff = hiHalf * 8;    // elements
  const unsigned short* aRow = A + (size_t)(rowBase + l15) * H_;
#pragma unroll 1
  for (int ks = 0; ks < H_; ks += 32) {
    FragBF a;
    a.q.lo = *(const uint4*)(aRow + ks + aKoff);        // K 0..7  (rel)
    a.q.hi = *(const uint4*)(aRow + ks + aKoff + 16);   // K 16..23 (rel)
#pragma unroll
    for (int g = 0; g < 4; ++g) {
      const int col = g * H_ + colBase + l15;           // gate column = B column
      const unsigned short* bp = W + (size_t)col * H_ + ks + bKoff;
      FragBF b;
      b.q.lo = *(const uint4*)bp;          // K 0..7  (rel half)
      b.q.hi = *(const uint4*)(bp + 16);   // K 16..23 (rel)
      acc[g] = __builtin_amdgcn_wmma_f32_16x16x32_bf16(
          false, a.v, false, b.v, (short)0, acc[g], false, false);
    }
  }
}

// ---------------- gate nonlinearity + h store ----------------
template <bool HASX>
__device__ __forceinline__ void update_store(const v8f* acc, float c[8],
                                             const float b[4], const float wx[4],
                                             const float xv[8],
                                             unsigned short* __restrict__ hOut,
                                             int rowBase, int colBase,
                                             int l15, int hiHalf) {
  const int col = colBase + l15;
#pragma unroll
  for (int e = 0; e < 8; ++e) {
    float gi = acc[0][e] + b[0];
    float gf = acc[1][e] + b[1];
    float gg = acc[2][e] + b[2];
    float go = acc[3][e] + b[3];
    if (HASX) {
      gi += xv[e] * wx[0];
      gf += xv[e] * wx[1];
      gg += xv[e] * wx[2];
      go += xv[e] * wx[3];
    }
    const float cn = sigmoidf_(gf) * c[e] + sigmoidf_(gi) * tanhf(gg);
    c[e] = cn;
    const float hv = sigmoidf_(go) * tanhf(cn);
    const int row = rowBase + e + 8 * hiHalf;        // C-layout: VGPR e, lane half
    hOut[(size_t)row * H_ + col] = f2bf(hv);
  }
}

// ---------------- persistent 2-layer LSTM ----------------
__global__ __launch_bounds__(TPB, 1)
void lstm_persistent_kernel(const float* __restrict__ y,        // [N_,L_]
                            const float* __restrict__ Wih1,     // [G4_] (4H x 1)
                            const float* __restrict__ bih1, const float* __restrict__ bhh1,
                            const float* __restrict__ bih2, const float* __restrict__ bhh2,
                            const float* __restrict__ Wlin,     // [H_]
                            const float* __restrict__ blin,     // [1]
                            const unsigned short* __restrict__ Whh1,  // bf16 [G4_,H_]
                            const unsigned short* __restrict__ Wih2,  // bf16 [G4_,H_]
                            const unsigned short* __restrict__ Whh2,  // bf16 [G4_,H_]
                            unsigned short* __restrict__ h1buf,       // bf16 [2][N_][H_]
                            unsigned short* __restrict__ h2buf,       // bf16 [2][N_][H_]
                            unsigned* __restrict__ barCnt,
                            float* __restrict__ out) {                // [N_,L_]
  const int wg      = blockIdx.x;        // 0..31
  const int tid     = threadIdx.x;
  const int wv      = tid >> 5;          // wave 0..7
  const int lane    = tid & 31;
  const int l15     = lane & 15;
  const int hiHalf  = lane >> 4;
  const int rowBase = wv * 16;           // batch rows this wave owns
  const int colBase = wg * WCOLS;        // H-columns this WG owns (16)

  // Per-lane gate biases, input-weight column (rank-1 x term)
  float bias1[4], bias2[4], wx1[4];
#pragma unroll
  for (int g = 0; g < 4; ++g) {
    const int col = g * H_ + colBase + l15;
    bias1[g] = bih1[col] + bhh1[col];
    bias2[g] = bih2[col] + bhh2[col];
    wx1[g]   = Wih1[col];
  }

  float c1[8], c2[8];
#pragma unroll
  for (int e = 0; e < 8; ++e) { c1[e] = 0.f; c2[e] = 0.f; }

  const float bl = blin[0];
  unsigned bar = 0;

  for (int t = 0; t < L_; ++t) {
    const unsigned short* h1r = h1buf + (size_t)(t & 1) * N_ * H_;
    unsigned short*       h1w = h1buf + (size_t)((t + 1) & 1) * N_ * H_;
    const unsigned short* h2r = h2buf + (size_t)(t & 1) * N_ * H_;
    unsigned short*       h2w = h2buf + (size_t)((t + 1) & 1) * N_ * H_;

    // ---- layer 1: gates1 = h1(t-1) @ Whh1^T (+ x*Wih1 + bias in update) ----
    v8f acc[4];
#pragma unroll
    for (int i = 0; i < 4; ++i) acc[i] = (v8f)(0.0f);
    gemm_acc(acc, h1r, Whh1, rowBase, colBase, l15, hiHalf);

    // x_t for this wave's rows (loaded late: keeps GEMM live range clean)
    float xv[8];
#pragma unroll
    for (int e = 0; e < 8; ++e) {
      const int row = rowBase + e + 8 * hiHalf;
      xv[e] = y[(size_t)row * L_ + t];
    }
    update_store<true>(acc, c1, bias1, wx1, xv, h1w, rowBase, colBase, l15, hiHalf);

    grid_barrier(barCnt, ++bar);   // h1(t) visible to all WGs

    // ---- layer 2: gates2 = h1(t) @ Wih2^T + h2(t-1) @ Whh2^T ----
#pragma unroll
    for (int i = 0; i < 4; ++i) acc[i] = (v8f)(0.0f);
    gemm_acc(acc, h1w, Wih2, rowBase, colBase, l15, hiHalf);
    gemm_acc(acc, h2r, Whh2, rowBase, colBase, l15, hiHalf);
    update_store<false>(acc, c2, bias2, wx1, xv, h2w, rowBase, colBase, l15, hiHalf);

    grid_barrier(barCnt, ++bar);   // h2(t) visible to all WGs

    // ---- output: out[n,t] = h2(t)[n,:] . Wlin + blin ; WG wg owns rows 4wg..4wg+3 ----
    if (wv < 4) {
      const int n = wg * 4 + wv;
      const unsigned short* hrow = h2w + (size_t)n * H_;
      const int k0 = lane * 16;
      float sum = 0.f;
#pragma unroll
      for (int j = 0; j < 16; ++j) sum += bf2f(hrow[k0 + j]) * Wlin[k0 + j];
#pragma unroll
      for (int off = 16; off >= 1; off >>= 1) sum += __shfl_xor(sum, off, 32);
      if (lane == 0) out[(size_t)n * L_ + t] = sum + bl;
    }
  }
}

// ---------------- host launch ----------------
extern "C" void kernel_launch(void* const* d_in, const int* in_sizes, int n_in,
                              void* d_out, int out_size, void* d_ws, size_t ws_size,
                              hipStream_t stream) {
  (void)in_sizes; (void)n_in; (void)out_size; (void)ws_size;

  const float* y     = (const float*)d_in[0];
  const float* Wih1  = (const float*)d_in[1];
  const float* Whh1f = (const float*)d_in[2];
  const float* bih1  = (const float*)d_in[3];
  const float* bhh1  = (const float*)d_in[4];
  const float* Wih2f = (const float*)d_in[5];
  const float* Whh2f = (const float*)d_in[6];
  const float* bih2  = (const float*)d_in[7];
  const float* bhh2  = (const float*)d_in[8];
  const float* Wlin  = (const float*)d_in[9];
  const float* blin  = (const float*)d_in[10];
  float* out = (float*)d_out;

  // Workspace carve-up (256B aligned): counter + 3 bf16 weight mats + 2 h ping-pong bufs
  char* ws = (char*)d_ws;
  size_t off = 0;
  auto alloc = [&](size_t bytes) -> void* {
    off = (off + 255) & ~(size_t)255;
    void* p = ws + off;
    off += bytes;
    return p;
  };
  unsigned*       cnt   = (unsigned*)alloc(256);
  unsigned short* whh1b = (unsigned short*)alloc((size_t)G4_ * H_ * 2);
  unsigned short* wih2b = (unsigned short*)alloc((size_t)G4_ * H_ * 2);
  unsigned short* whh2b = (unsigned short*)alloc((size_t)G4_ * H_ * 2);
  unsigned short* h1b   = (unsigned short*)alloc((size_t)2 * N_ * H_ * 2);
  unsigned short* h2b   = (unsigned short*)alloc((size_t)2 * N_ * H_ * 2);

  const int nw = G4_ * H_;                    // 1048576 elements per weight matrix
  const int cvtBlocks = (nw + TPB - 1) / TPB;
  cvt_bf16_kernel<<<cvtBlocks, TPB, 0, stream>>>(Whh1f, whh1b, nw);
  cvt_bf16_kernel<<<cvtBlocks, TPB, 0, stream>>>(Wih2f, wih2b, nw);
  cvt_bf16_kernel<<<cvtBlocks, TPB, 0, stream>>>(Whh2f, whh2b, nw);

  const int ni = 2 * N_ * H_;
  init_state_kernel<<<(ni + TPB - 1) / TPB, TPB, 0, stream>>>(h1b, h2b, cnt);

  lstm_persistent_kernel<<<NWG, TPB, 0, stream>>>(
      y, Wih1, bih1, bhh1, bih2, bhh2, Wlin, blin,
      whh1b, wih2b, whh2b, h1b, h2b, cnt, out);
}
